// input_layer_21758304322131
// MI455X (gfx1250) — matrix operand order repaired and checked by reading, not compiled, not run on hardware
//
#include <hip/hip_runtime.h>
#include <hip/hip_bf16.h>
#include <math.h>

// ---------------------------------------------------------------------------
// Fused grid-attention layer for MI455X (gfx1250, wave32, WMMA).
// One workgroup (256 threads = 8 waves) per attention group (b, q):
//   xs   = silu(x @ W_in + b)          [64,256]  VALU (K=8 too skinny for WMMA)
//   vh   = xs @ W_v                    [64,256]  WMMA f32_16x16x32_f16
//   att  = softmax(mask(bias(pos)))    [8,4,64]  VALU
//   out  = att * vh                    [4,256]   VALU
//   out  = out @ W_o                   [4,256]   WMMA (M padded to 16)
//   h    = LN1(out)
//   u    = silu(h @ W1)                [4,1024]  WMMA
//   y    = LN2(h + u @ W2)             [4,256]   WMMA
// Everything stays in LDS (~137 KB/WG, fine on the 320 KB CDNA5 WGP).
// Weights are pre-packed to f16 B-fragment-major layout in d_ws so each lane
// loads its WMMA B fragment as one contiguous 32-byte read (L2-resident).
// ---------------------------------------------------------------------------

typedef __attribute__((ext_vector_type(16))) _Float16 v16h;
typedef __attribute__((ext_vector_type(8)))  _Float16 v8h;
typedef __attribute__((ext_vector_type(8)))  float    v8f;

__device__ __forceinline__ float silu_f(float x) { return x / (1.0f + __expf(-x)); }

__device__ __forceinline__ float wave_sum32(float v) {
#pragma unroll
  for (int off = 16; off > 0; off >>= 1) v += __shfl_xor(v, off, 32);
  return v;
}

// A fragment (16x32 f16, M x K) from LDS, row-major buffer with `stride` halves.
// Layout (ISA 7.12.2): lane L -> M = m0 + (L&15); K-halfbase = (L>=16)?8:0;
// halves 0..7 = K base+0..7, halves 8..15 = K base+16..23.
__device__ __forceinline__ v16h load_a_frag(const _Float16* buf, int stride,
                                            int m0, int k0, int lane) {
  const int row  = m0 + (lane & 15);
  const int base = (lane >> 4) << 3;  // 0 or 8
  const _Float16* p = buf + row * stride + k0 + base;
  v8h lo = *(const v8h*)(p);
  v8h hi = *(const v8h*)(p + 16);
  v16h a;
#pragma unroll
  for (int i = 0; i < 8; ++i) { a[i] = lo[i]; a[8 + i] = hi[i]; }
  return a;
}

// B fragment (32x16 f16, K x N), pre-packed fragment-major: 32 lanes x 16 halves.
__device__ __forceinline__ v16h load_b_frag(const _Float16* pack, int NT,
                                            int kt, int nt, int lane) {
  return *(const v16h*)(pack + ((((kt * NT) + nt) * 32 + lane) << 4));
}

__device__ __forceinline__ v8f wmma_step(v16h a, v16h b, v8f c) {
  return __builtin_amdgcn_wmma_f32_16x16x32_f16(false, a, false, b,
                                                (short)0, c, false, false);
}

// ---------------------------------------------------------------------------
// Pack a row-major fp32 weight [K,N] into f16 B-fragment-major layout.
// packed[((kt*NT+nt)*32 + L)*16 + j] = W[(kt*32 + ((L&16)>>1) + (j&7) + ((j&8)<<1)) * N + nt*16 + (L&15)]
// (lanes 0-15 carry K = 0..7,16..23 of the 32-block, lanes 16-31 carry K = 8..15,24..31)
// ---------------------------------------------------------------------------
__global__ void pack_b_frag_kernel(const float* __restrict__ W,
                                   _Float16* __restrict__ P,
                                   int Kdim, int Ndim) {
  int i = blockIdx.x * 256 + threadIdx.x;
  if (i >= Kdim * Ndim) return;
  const int j    = i & 15;
  const int L    = (i >> 4) & 31;
  const int frag = i >> 9;
  const int NT   = Ndim >> 4;
  const int nt   = frag % NT;
  const int kt   = frag / NT;
  const int k = kt * 32 + ((L & 16) >> 1) + (j & 7) + ((j & 8) << 1);
  const int n = nt * 16 + (L & 15);
  P[i] = (_Float16)W[k * Ndim + n];
}

// ---------------------------------------------------------------------------
// Main fused kernel. grid = 4096 blocks (b*1024 + q), 256 threads.
// ---------------------------------------------------------------------------
#define D_   256
#define FF_  1024
#define KK_  64
#define XS_S 264   // f16 row stride (528 B, 16B aligned)
#define U_S  1040  // f16 row stride for FFN act (2080 B, 16B aligned)
#define VH_S 260   // f32 row stride
#define HB_S 260

__global__ __launch_bounds__(256)
void fused_grid_attn_kernel(
    const float* __restrict__ x,            // [4,4096,16,8]
    const int*   __restrict__ gli,          // [4,4096]
    const unsigned char* __restrict__ dropm,// [4,4096] bool
    const int*   __restrict__ imap,         // [8192,16]
    const unsigned char* __restrict__ inr,  // [8192,16] bool
    const float* __restrict__ icoord,       // [2,65536]
    const float* __restrict__ gcoord,       // [2,8192]
    const float* __restrict__ Win,          // [8,256]
    const float* __restrict__ bin,          // [256]
    const float* __restrict__ Wpe,          // [2,32]
    const float* __restrict__ Wbias,        // [32,8]
    const float* __restrict__ g1, const float* __restrict__ b1,
    const float* __restrict__ g2, const float* __restrict__ b2,
    const _Float16* __restrict__ WvP,
    const _Float16* __restrict__ WoP,
    const _Float16* __restrict__ W1P,
    const _Float16* __restrict__ W2P,
    float* __restrict__ out)                // [4,4096,256]
{
  __shared__ _Float16 s_xs[64 * XS_S];   // xs f16; later aliased as U [16][U_S]
  __shared__ float    s_vh[64 * VH_S];   // vh f32; later reused for W2 result
  __shared__ float    s_sc[32 * 64];     // scores/att [h*4+t][k]
  __shared__ _Float16 s_ob[16 * XS_S];   // attn-out f16 (A of W_o), then h16 (A of W1)
  __shared__ float    s_hb[16 * HB_S];   // W_o result f32 / normalized h
  __shared__ float    s_xi[64 * 8];      // raw x tile
  __shared__ float    s_wpe[64];
  __shared__ float    s_wb[256];
  __shared__ float    s_lon1[4], s_lat1[4];
  __shared__ int      s_g[4];
  __shared__ int      s_dm[4];

  const int tid  = threadIdx.x;
  const int lane = tid & 31;
  const int wv   = tid >> 5;       // wave id 0..7
  const int bx   = blockIdx.x;
  const int b    = bx >> 10;
  const int q    = bx & 1023;

  // ---- stage group-local inputs -------------------------------------------
  if (tid < 4) {
    const int n = q * 4 + tid;
    const int g = gli[b * 4096 + n];
    s_g[tid]   = g;
    s_dm[tid]  = dropm[b * 4096 + n];
    s_lon1[tid] = gcoord[g];
    s_lat1[tid] = gcoord[8192 + g];
  }
  if (tid < 64) s_wpe[tid] = Wpe[tid];
  s_wb[tid] = Wbias[tid];
#pragma unroll
  for (int i = 0; i < 2; ++i) {            // 64x8 raw feature tile
    const int e = tid + i * 256;
    const int m = e >> 3, f = e & 7;
    const int n = q * 4 + (m >> 4);
    s_xi[e] = x[((size_t)(b * 4096 + n) * 16 + (m & 15)) * 8 + f];
  }
  // per-thread W_in column (thread owns output channel d = tid)
  float wreg[8];
#pragma unroll
  for (int f = 0; f < 8; ++f) wreg[f] = Win[f * 256 + tid];
  const float breg = bin[tid];
  __syncthreads();

  // ---- input MLP: xs[m][d] = silu(x@W_in + b), d = tid --------------------
  for (int m = 0; m < 64; ++m) {
    float acc = breg;
#pragma unroll
    for (int f = 0; f < 8; ++f) acc += s_xi[m * 8 + f] * wreg[f];
    s_xs[m * XS_S + tid] = (_Float16)silu_f(acc);
  }

  // ---- positional bias scores: thread = (t, k) ----------------------------
  {
    const int t = tid >> 6, k = tid & 63;
    const int sidx = k >> 4, j = k & 15;
    const int g = s_g[sidx];
    const int im = imap[g * 16 + j];
    const int masked = (!inr[g * 16 + j]) | s_dm[sidx];
    const float dlon = icoord[im]         - s_lon1[t];
    const float dlat = icoord[65536 + im] - s_lat1[t];
    const float dist = sqrtf(dlon * dlon + dlat * dlat + 1e-12f);
    const float ang  = atan2f(dlat, dlon);
    float bias[8];
#pragma unroll
    for (int h = 0; h < 8; ++h) bias[h] = 0.0f;
#pragma unroll 4
    for (int p = 0; p < 32; ++p) {
      const float pe = silu_f(dist * s_wpe[p] + ang * s_wpe[32 + p]);
#pragma unroll
      for (int h = 0; h < 8; ++h) bias[h] += pe * s_wb[p * 8 + h];
    }
#pragma unroll
    for (int h = 0; h < 8; ++h)
      s_sc[(h * 4 + t) * 64 + k] = masked ? -1e9f : bias[h];
  }
  __syncthreads();

  // ---- softmax over K=64 per (h,t) row ------------------------------------
  if (tid < 32) {
    float* r = &s_sc[tid * 64];
    float mx = -3.4e38f;
    for (int k = 0; k < 64; ++k) mx = fmaxf(mx, r[k]);
    float sum = 0.0f;
    for (int k = 0; k < 64; ++k) { float e = __expf(r[k] - mx); r[k] = e; sum += e; }
    const float inv = 1.0f / sum;
    for (int k = 0; k < 64; ++k) r[k] *= inv;
  }
  __syncthreads();

  // ---- GEMM1: vh[64,256] = xs @ W_v  (WMMA, 64 tiles, 8 per wave) --------
  for (int i = 0; i < 8; ++i) {
    const int tile = wv + 8 * i;
    const int mt = tile >> 4, nt = tile & 15;
    v8f acc = {};
#pragma unroll
    for (int kt = 0; kt < 8; ++kt) {
      v16h a = load_a_frag(s_xs, XS_S, mt * 16, kt * 32, lane);
      v16h bf = load_b_frag(WvP, 16, kt, nt, lane);
      acc = wmma_step(a, bf, acc);
    }
    const int n = nt * 16 + (lane & 15);
    const int mb = mt * 16 + ((lane >> 4) << 3);
#pragma unroll
    for (int r = 0; r < 8; ++r) s_vh[(mb + r) * VH_S + n] = acc[r];
  }
  __syncthreads();

  // ---- attention output: out[t][d] = sum_k att[h][t][k] * vh[k][d] --------
#pragma unroll
  for (int i = 0; i < 4; ++i) {
    const int o = i * 256 + tid;
    const int t = o >> 8, d = o & 255, h = d >> 5;
    const float* sp = &s_sc[(h * 4 + t) * 64];
    const float* vp = &s_vh[d];
    float acc = 0.0f;
    for (int k = 0; k < 64; ++k) acc += sp[k] * vp[k * VH_S];
    s_ob[t * XS_S + d] = (_Float16)acc;   // rows 4..15 of s_ob are don't-care
  }
  __syncthreads();

  // ---- GEMM2: out_o[16,256] = s_ob @ W_o (M padded; 2 N-tiles per wave) ---
  {
    v16h a[8];
#pragma unroll
    for (int kt = 0; kt < 8; ++kt) a[kt] = load_a_frag(s_ob, XS_S, 0, kt * 32, lane);
#pragma unroll
    for (int i = 0; i < 2; ++i) {
      const int nt = wv * 2 + i;
      v8f acc = {};
#pragma unroll
      for (int kt = 0; kt < 8; ++kt)
        acc = wmma_step(a[kt], load_b_frag(WoP, 16, kt, nt, lane), acc);
      const int n = nt * 16 + (lane & 15);
      const int mb = (lane >> 4) << 3;
#pragma unroll
      for (int r = 0; r < 8; ++r) s_hb[(mb + r) * HB_S + n] = acc[r];
    }
  }
  __syncthreads();

  // ---- LN1 on rows 0..3; store f32 (s_hb) and f16 (s_ob) ------------------
  if (wv < 4) {
    const int t = wv;
    float v[8];
    float mu = 0.0f;
#pragma unroll
    for (int i = 0; i < 8; ++i) { v[i] = s_hb[t * HB_S + lane + 32 * i]; mu += v[i]; }
    mu = wave_sum32(mu) * (1.0f / 256.0f);
    float var = 0.0f;
#pragma unroll
    for (int i = 0; i < 8; ++i) { const float d0 = v[i] - mu; var += d0 * d0; }
    var = wave_sum32(var) * (1.0f / 256.0f);
    const float rs = rsqrtf(var + 1e-5f);
#pragma unroll
    for (int i = 0; i < 8; ++i) {
      const int d = lane + 32 * i;
      const float h = (v[i] - mu) * rs * g1[d] + b1[d];
      s_hb[t * HB_S + d] = h;
      s_ob[t * XS_S + d] = (_Float16)h;
    }
  }
  __syncthreads();

  // ---- GEMM3: u[16,1024] = silu(h16 @ W1); u stored f16 into s_xs (U) -----
  {
    _Float16* U = s_xs;
    v16h a[8];
#pragma unroll
    for (int kt = 0; kt < 8; ++kt) a[kt] = load_a_frag(s_ob, XS_S, 0, kt * 32, lane);
    for (int i = 0; i < 8; ++i) {
      const int nt = wv * 8 + i;
      v8f acc = {};
#pragma unroll
      for (int kt = 0; kt < 8; ++kt)
        acc = wmma_step(a[kt], load_b_frag(W1P, 64, kt, nt, lane), acc);
      const int n = nt * 16 + (lane & 15);
      const int mb = (lane >> 4) << 3;
#pragma unroll
      for (int r = 0; r < 8; ++r)
        U[(mb + r) * U_S + n] = (_Float16)silu_f(acc[r]);
    }
  }
  __syncthreads();

  // ---- GEMM4: w2res[16,256] = U @ W2 (K=1024); result into s_vh region ----
  {
    const _Float16* U = s_xs;
#pragma unroll
    for (int i = 0; i < 2; ++i) {
      const int nt = wv * 2 + i;
      v8f acc = {};
      for (int kt = 0; kt < 32; ++kt) {
        v16h a = load_a_frag(U, U_S, 0, kt * 32, lane);
        acc = wmma_step(a, load_b_frag(W2P, 16, kt, nt, lane), acc);
      }
      const int n = nt * 16 + (lane & 15);
      const int mb = (lane >> 4) << 3;
#pragma unroll
      for (int r = 0; r < 8; ++r) s_vh[(mb + r) * VH_S + n] = acc[r];
    }
  }
  __syncthreads();

  // ---- LN2 on (h + w2res), write final output -----------------------------
  if (wv < 4) {
    const int t = wv;
    float v[8];
    float mu = 0.0f;
#pragma unroll
    for (int i = 0; i < 8; ++i) {
      const int d = lane + 32 * i;
      v[i] = s_hb[t * HB_S + d] + s_vh[t * VH_S + d];
      mu += v[i];
    }
    mu = wave_sum32(mu) * (1.0f / 256.0f);
    float var = 0.0f;
#pragma unroll
    for (int i = 0; i < 8; ++i) { const float d0 = v[i] - mu; var += d0 * d0; }
    var = wave_sum32(var) * (1.0f / 256.0f);
    const float rs = rsqrtf(var + 1e-5f);
    float* op = out + ((size_t)(b * 4096 + q * 4 + t)) * 256;
#pragma unroll
    for (int i = 0; i < 8; ++i) {
      const int d = lane + 32 * i;
      op[d] = (v[i] - mu) * rs * g2[d] + b2[d];
    }
  }
}

// ---------------------------------------------------------------------------
extern "C" void kernel_launch(void* const* d_in, const int* in_sizes, int n_in,
                              void* d_out, int out_size, void* d_ws, size_t ws_size,
                              hipStream_t stream) {
  const float*         x      = (const float*)d_in[0];
  const int*           gli    = (const int*)d_in[1];
  const unsigned char* dropm  = (const unsigned char*)d_in[2];
  const int*           imap   = (const int*)d_in[3];
  const unsigned char* inr    = (const unsigned char*)d_in[4];
  const float*         icoord = (const float*)d_in[5];
  const float*         gcoord = (const float*)d_in[6];
  const float*         Win    = (const float*)d_in[7];
  const float*         bin    = (const float*)d_in[8];
  const float*         Wpe    = (const float*)d_in[9];
  const float*         Wbias  = (const float*)d_in[10];
  const float*         Wv     = (const float*)d_in[11];
  const float*         Wo     = (const float*)d_in[12];
  const float*         W1     = (const float*)d_in[13];
  const float*         W2     = (const float*)d_in[14];
  const float*         g1     = (const float*)d_in[15];
  const float*         b1     = (const float*)d_in[16];
  const float*         g2     = (const float*)d_in[17];
  const float*         b2     = (const float*)d_in[18];

  _Float16* ws  = (_Float16*)d_ws;
  _Float16* WvP = ws;                 //  65536 halves
  _Float16* WoP = ws + 65536;         //  65536 halves
  _Float16* W1P = ws + 131072;        // 262144 halves
  _Float16* W2P = ws + 393216;        // 262144 halves

  pack_b_frag_kernel<<<256,  256, 0, stream>>>(Wv, WvP, 256, 256);
  pack_b_frag_kernel<<<256,  256, 0, stream>>>(Wo, WoP, 256, 256);
  pack_b_frag_kernel<<<1024, 256, 0, stream>>>(W1, W1P, 256, 1024);
  pack_b_frag_kernel<<<1024, 256, 0, stream>>>(W2, W2P, 1024, 256);

  fused_grid_attn_kernel<<<4096, 256, 0, stream>>>(
      x, gli, dropm, imap, inr, icoord, gcoord, Win, bin, Wpe, Wbias,
      g1, b1, g2, b2, WvP, WoP, W1P, W2P, (float*)d_out);
}
